// RecurrentLIFReservoir_28922309771719
// MI455X (gfx1250) — matrix operand order repaired and checked by reading, not compiled, not run on hardware
//
#include <hip/hip_runtime.h>
#include <hip/hip_bf16.h>

typedef __attribute__((ext_vector_type(16))) _Float16 v16h;
typedef __attribute__((ext_vector_type(8)))  float    v8f;

#define B_  16
#define C_  256
#define T_  512
#define H_  1024
#define ALPHA_V 0.95122942450071400f   // exp(-0.001/0.02)
#define ALPHA_S 0.90483741803595952f   // exp(-0.001/0.01)
#define V_TH_   1.0f

#define WMMA16(A, Bm, Cm) __builtin_amdgcn_wmma_f32_16x16x32_f16( \
    false, (A), false, (Bm), (short)0, (Cm), false, false)

// ---- WMMA fragment offset helpers (within one 16x32 / 32x16 f16 tile = 512 halfs) ----
// A (16xK=32, f16): lanes 0-15 hold K={0..7,16..23}, lanes 16-31 hold K={8..15,24..31},
// 2 halfs per VGPR, 8 VGPRs = 16 halfs per lane, stored contiguously per lane.
__device__ __forceinline__ int a_off(int m, int ko) {
    int lane = m + ((ko & 8) ? 16 : 0);
    int j = ((ko >> 4) << 2) + ((ko & 7) >> 1);
    return lane * 16 + (j << 1) + (ko & 1);
}
// B (K=32 x 16, f16): lanes 0-15 hold K={0..7,16..23} (col N=lane), lanes 16-31 hold K={8..15,24..31}.
__device__ __forceinline__ int b_off(int n, int ko) {
    int lane = n + ((ko & 8) ? 16 : 0);
    int j = ((ko >> 4) << 2) + ((ko & 7) >> 1);
    return lane * 16 + (j << 1) + (ko & 1);
}

// ---------------- one-time convert / swizzle kernels ----------------
// w_in (H x C) f32 row-major -> A-fragment tiles (mt over H/16, kt over C/32)
__global__ void conv_win(const float* __restrict__ w_in, _Float16* __restrict__ wA) {
    int idx = blockIdx.x * blockDim.x + threadIdx.x;   // h*C + c
    int h = idx >> 8, c = idx & (C_ - 1);
    int tile = (h >> 4) * (C_ / 32) + (c >> 5);
    wA[tile * 512 + a_off(h & 15, c & 31)] = (_Float16)w_in[idx];
}
// s_in (b, C, T) f32 -> B-fragment tiles per batch: (nt over T/16, kt over C/32)
__global__ void conv_sin(const float* __restrict__ s_in, _Float16* __restrict__ sB) {
    int idx = blockIdx.x * blockDim.x + threadIdx.x;   // b*C*T + c*T + t
    int b = idx >> 17;
    int rem = idx & (C_ * T_ - 1);
    int c = rem >> 9, t = rem & (T_ - 1);
    int tile = (b * (T_ / 16) + (t >> 4)) * (C_ / 32) + (c >> 5);
    sB[tile * 512 + b_off(t & 15, c & 31)] = (_Float16)s_in[idx];
}
// w_rec (H x H) f32 row-major; B[k][n] = w_rec[n][k] -> B-fragment tiles (nt over H/16, kt over H/32)
__global__ void conv_wrec(const float* __restrict__ w_rec, _Float16* __restrict__ wB) {
    int idx = blockIdx.x * blockDim.x + threadIdx.x;   // n*H + k
    int n = idx >> 10, k = idx & (H_ - 1);
    int tile = (n >> 4) * (H_ / 32) + (k >> 5);
    wB[tile * 512 + b_off(n & 15, k & 31)] = (_Float16)w_rec[idx];
}
// zero membrane/current state and both spike fragment buffers (16384 elements each)
__global__ void init_state(float* __restrict__ v_st, float* __restrict__ i_st,
                           _Float16* __restrict__ sp0, _Float16* __restrict__ sp1) {
    int i = blockIdx.x * blockDim.x + threadIdx.x;
    v_st[i] = 0.0f; i_st[i] = 0.0f;
    sp0[i] = (_Float16)0.0f; sp1[i] = (_Float16)0.0f;
}

// ---------------- phase 1: I_in[b,h,t] = w_in @ s_in[b]  -> out_v half of d_out ----------------
__global__ void gemm_input_drive(const v16h* __restrict__ wA, const v16h* __restrict__ sB,
                                 float* __restrict__ out_v) {
    int gwave = (blockIdx.x * blockDim.x + threadIdx.x) >> 5;
    int lane  = threadIdx.x & 31;
    int nt = gwave & 31;           // T tile
    int mt = (gwave >> 5) & 63;    // H tile
    int b  = gwave >> 11;          // batch
    v8f acc = {};
#pragma unroll
    for (int kt = 0; kt < C_ / 32; ++kt) {
        v16h a  = wA[(mt * (C_ / 32) + kt) * 32 + lane];
        v16h bb = sB[((b * (T_ / 16) + nt) * (C_ / 32) + kt) * 32 + lane];
        acc = WMMA16(a, bb, acc);
    }
    int t = nt * 16 + (lane & 15);
    int mbase = (lane < 16) ? 0 : 8;
#pragma unroll
    for (int i = 0; i < 8; ++i) {
        int h = mt * 16 + mbase + i;
        out_v[((size_t)b * H_ + h) * T_ + t] = acc[i];
    }
}

// ---------------- phase 2: one recurrent LIF step ----------------
// grid = 64 blocks (one 16-neuron tile each), 256 threads = 8 waves; K=1024 split 8-ways.
__global__ void lif_step(const v16h* __restrict__ spA, const v16h* __restrict__ wB,
                         _Float16* __restrict__ spA_next,
                         float* __restrict__ v_st, float* __restrict__ i_st,
                         const float* __restrict__ bias,
                         float* __restrict__ out_sp, float* __restrict__ out_v, int t) {
    __shared__ float red[8 * 256];
    int lane = threadIdx.x & 31;
    int w    = threadIdx.x >> 5;
    int nt   = blockIdx.x;
    v8f acc = {};
#pragma unroll
    for (int kk = 0; kk < 4; ++kk) {
        int kt = w * 4 + kk;
        v16h a  = spA[kt * 32 + lane];
        v16h bb = wB[(nt * (H_ / 32) + kt) * 32 + lane];
        acc = WMMA16(a, bb, acc);
    }
#pragma unroll
    for (int i = 0; i < 8; ++i) red[w * 256 + i * 32 + lane] = acc[i];
    __syncthreads();

    int tid = threadIdx.x;              // tid = i*32 + l
    float r = 0.0f;
#pragma unroll
    for (int ww = 0; ww < 8; ++ww) r += red[ww * 256 + tid];

    int l = tid & 31, i = tid >> 5;
    int b = i + ((l >= 16) ? 8 : 0);    // batch (M)
    int h = nt * 16 + (l & 15);         // neuron (N)
    size_t st = (size_t)b * H_ + h;
    size_t ot = st * T_ + t;

    float iin = out_v[ot];              // I_in stashed in v half of d_out
    float Inew = ALPHA_S * i_st[st] + iin + r;
    float v = ALPHA_V * v_st[st] + Inew + bias[h];
    float s = (v > V_TH_) ? 1.0f : 0.0f;
    float vout = v - s * V_TH_;

    out_sp[ot] = s;
    out_v[ot]  = vout;
    i_st[st] = Inew;
    v_st[st] = vout;
    // write next-step spike directly in A-fragment order
    int kt2 = h >> 5, ko = h & 31;
    spA_next[kt2 * 512 + a_off(b, ko)] = (_Float16)s;
}

extern "C" void kernel_launch(void* const* d_in, const int* in_sizes, int n_in,
                              void* d_out, int out_size, void* d_ws, size_t ws_size,
                              hipStream_t stream) {
    const float* s_in  = (const float*)d_in[0];   // (16,256,512)
    const float* w_in  = (const float*)d_in[1];   // (1024,256)
    const float* w_rec = (const float*)d_in[2];   // (1024,1024)
    const float* bias  = (const float*)d_in[3];   // (1024,)
    float* out_sp = (float*)d_out;
    float* out_v  = out_sp + (size_t)B_ * H_ * T_;

    char* ws = (char*)d_ws;
    _Float16* wA  = (_Float16*)(ws + 0);                 // 512 KB
    _Float16* sB  = (_Float16*)(ws + 524288);            // 4 MB
    _Float16* wB  = (_Float16*)(ws + 4718592);           // 2 MB
    _Float16* sp0 = (_Float16*)(ws + 6815744);           // 32 KB
    _Float16* sp1 = (_Float16*)(ws + 6848512);           // 32 KB
    float*    vst = (float*)(ws + 6881280);              // 64 KB
    float*    ist = (float*)(ws + 6946816);              // 64 KB

    conv_win <<<(H_ * C_) / 256, 256, 0, stream>>>(w_in, wA);
    conv_sin <<<(B_ * C_ * T_) / 256, 256, 0, stream>>>(s_in, sB);
    conv_wrec<<<(H_ * H_) / 256, 256, 0, stream>>>(w_rec, wB);
    init_state<<<(B_ * H_) / 256, 256, 0, stream>>>(vst, ist, sp0, sp1);

    // phase 1: 16 * 64 * 32 wave-tiles, 8 waves per block
    gemm_input_drive<<<(B_ * 64 * 32) / 8, 256, 0, stream>>>((const v16h*)wA, (const v16h*)sB, out_v);

    // phase 2: 512 dependent steps, ping-pong spike fragment buffers
    for (int t = 0; t < T_; ++t) {
        _Float16* sprev = (t & 1) ? sp1 : sp0;
        _Float16* snext = (t & 1) ? sp0 : sp1;
        lif_step<<<H_ / 16, 256, 0, stream>>>((const v16h*)sprev, (const v16h*)wB, snext,
                                              vst, ist, bias, out_sp, out_v, t);
    }
}
